// GBM_68539088109972
// MI455X (gfx1250) — compile-verified, matrix operand order checked by reading, not run on hardware
//
#include <hip/hip_runtime.h>
#include <hip/hip_bf16.h>
#include <math.h>

// ---------------------------------------------------------------------------
// Shapes (compile-time constants from the reference)
// ---------------------------------------------------------------------------
#define BATCH   2
#define SEQL    256
#define FLATHW  32768
#define DMODEL  1024
#define DINNER  2048
#define DSTATE  16
#define DTRANK  64

// ---------------------------------------------------------------------------
// WMMA GEMM tiling
// ---------------------------------------------------------------------------
#define TM  128        // workgroup M tile
#define TN  128        // workgroup N tile
#define KS  64         // K step staged per iteration
#define KSP 72         // padded LDS row stride (bf16 elems) to spread banks

typedef __bf16 bf16;
typedef __attribute__((ext_vector_type(16))) __bf16 v16bf;
typedef __attribute__((ext_vector_type(8)))  float  v8f;

union FragU { uint4 q[2]; v16bf v; };

__device__ __forceinline__ unsigned pack2_bf16(float a, float b) {
    union { bf16 h[2]; unsigned u; } t;
    t.h[0] = (bf16)a; t.h[1] = (bf16)b;
    return t.u;
}

__device__ __forceinline__ float act_apply(float v, int act) {
    if (act == 1) return 1.0f / (1.0f + __expf(-v));               // sigmoid
    if (act == 2) return (v > 20.0f) ? v : log1pf(__expf(v));      // softplus
    return v;
}

// ---- stage a 128x64 f32 slab of A and W into bf16 register packs ----------
// Branch-free when GUARD_N == false (N multiple of TN): 8x global_load_b128
// per matrix per thread, fully unrolled.
template <bool GUARD_N>
__device__ __forceinline__ void stage_load(const float* __restrict__ A, int lda,
                                           const float* __restrict__ W,
                                           int K, int N, int blockM, int blockN,
                                           int k0, int tid,
                                           unsigned* a_pk, unsigned* b_pk)
{
    #pragma unroll
    for (int i = 0; i < 8; ++i) {
        int q   = tid + i * 256;      // quad id 0..2047 of the 128x64 tile
        int row = q >> 4;             // 16 quads per row
        int c4  = (q & 15) * 4;
        float4 fa = *(const float4*)(A + (size_t)(blockM + row) * lda + k0 + c4);
        a_pk[2 * i]     = pack2_bf16(fa.x, fa.y);
        a_pk[2 * i + 1] = pack2_bf16(fa.z, fa.w);
        float4 fb = make_float4(0.0f, 0.0f, 0.0f, 0.0f);
        int nrow = blockN + row;
        if (!GUARD_N || nrow < N)
            fb = *(const float4*)(W + (size_t)nrow * K + k0 + c4);
        b_pk[2 * i]     = pack2_bf16(fb.x, fb.y);
        b_pk[2 * i + 1] = pack2_bf16(fb.z, fb.w);
    }
}

__device__ __forceinline__ void stage_store(bf16* As, bf16* Bs, int tid,
                                            const unsigned* a_pk,
                                            const unsigned* b_pk)
{
    #pragma unroll
    for (int i = 0; i < 8; ++i) {
        int q   = tid + i * 256;
        int row = q >> 4;
        int c4  = (q & 15) * 4;
        *(uint2*)&As[row * KSP + c4] = make_uint2(a_pk[2 * i], a_pk[2 * i + 1]);
        *(uint2*)&Bs[row * KSP + c4] = make_uint2(b_pk[2 * i], b_pk[2 * i + 1]);
    }
}

// C[M,N] = act( A[M,K(lda)] * W[N,K]^T + bias + residual )
// slice==1: decoder epilogue — drop rows with l==SEQL-1, write to
//           out[(b*(SEQL-1)+l)*N + col]
template <bool GUARD_N>
__global__ __launch_bounds__(256)
void gemm_bf16_wmma(const float* __restrict__ A, int lda,
                    const float* __restrict__ W,
                    const float* __restrict__ bias,
                    const float* __restrict__ residual,
                    float* __restrict__ C,
                    int M, int N, int K,
                    int act, int slice)
{
    __shared__ __align__(16) bf16 As[TM * KSP];
    __shared__ __align__(16) bf16 Bs[TN * KSP];

    const int tid  = threadIdx.x;
    const int lane = tid & 31;
    const int wave = tid >> 5;
    const int wm   = (wave & 3) * 32;   // 4 waves down
    const int wn   = (wave >> 2) * 64;  // 2 waves across
    const int hi   = lane >> 4;         // half-wave: lanes 16-31 hold K+8
    const int l16  = lane & 15;

    // grid: x = M tiles (fast-varying -> M-blocks sharing a weight slab are
    // dispatched adjacently for L2 reuse), y = N tiles
    const int blockM = blockIdx.x * TM;
    const int blockN = blockIdx.y * TN;

    v8f acc[2][4];
    #pragma unroll
    for (int i = 0; i < 2; ++i)
        #pragma unroll
        for (int j = 0; j < 4; ++j)
            #pragma unroll
            for (int e = 0; e < 8; ++e)
                acc[i][j][e] = 0.0f;

    unsigned a_pk[16], b_pk[16];
    stage_load<GUARD_N>(A, lda, W, K, N, blockM, blockN, 0, tid, a_pk, b_pk);

    for (int k0 = 0; k0 < K; k0 += KS) {
        __syncthreads();                       // previous tile fully consumed
        stage_store(As, Bs, tid, a_pk, b_pk);  // publish current tile
        __syncthreads();

        // issue next slab's global loads before computing (latency overlap)
        if (k0 + KS < K)
            stage_load<GUARD_N>(A, lda, W, K, N, blockM, blockN, k0 + KS,
                                tid, a_pk, b_pk);
        // HBM prefetch two slabs ahead (speculative global_prefetch_b8)
        if (k0 + 2 * KS < K) {
            int nrow = blockN + (tid & 127);
            if (!GUARD_N || nrow < N)
                __builtin_prefetch(W + (size_t)nrow * K + k0 + 2 * KS, 0, 3);
            __builtin_prefetch(A + (size_t)(blockM + (tid & 127)) * lda
                                 + k0 + 2 * KS, 0, 3);
        }

        // ---- two 16x16x32 WMMA K-chunks per staged K-step ----
        #pragma unroll
        for (int kc = 0; kc < KS; kc += 32) {
            const int kb = kc + hi * 8;
            v16bf af[2], bfr[4];
            #pragma unroll
            for (int i = 0; i < 2; ++i) {
                int row = wm + 16 * i + l16;
                FragU f;
                f.q[0] = *(const uint4*)&As[row * KSP + kb];       // K kb..kb+7
                f.q[1] = *(const uint4*)&As[row * KSP + kb + 16];  // K kb+16..+23
                af[i] = f.v;
            }
            #pragma unroll
            for (int j = 0; j < 4; ++j) {
                int row = wn + 16 * j + l16;
                FragU f;
                f.q[0] = *(const uint4*)&Bs[row * KSP + kb];
                f.q[1] = *(const uint4*)&Bs[row * KSP + kb + 16];
                bfr[j] = f.v;
            }
            #pragma unroll
            for (int i = 0; i < 2; ++i)
                #pragma unroll
                for (int j = 0; j < 4; ++j)
                    acc[i][j] = __builtin_amdgcn_wmma_f32_16x16x32_bf16(
                        false, af[i], false, bfr[j], (short)0, acc[i][j],
                        false, false);
        }
    }

    // ---- fused epilogue: bias / residual / activation / slice-store ----
    #pragma unroll
    for (int i = 0; i < 2; ++i) {
        #pragma unroll
        for (int j = 0; j < 4; ++j) {
            #pragma unroll
            for (int v = 0; v < 8; ++v) {
                int row = blockM + wm + 16 * i + v + 8 * hi;  // ISA C layout
                int col = blockN + wn + 16 * j + l16;
                if (GUARD_N && col >= N) continue;
                float val = acc[i][j][v];
                if (bias)     val += bias[col];
                if (residual) val += residual[(size_t)row * N + col];
                val = act_apply(val, act);
                if (slice) {
                    int b = row >> 8, l = row & (SEQL - 1);
                    if (l == SEQL - 1) continue;
                    C[((size_t)(b * (SEQL - 1) + l)) * N + col] = val;
                } else {
                    C[(size_t)row * N + col] = val;
                }
            }
        }
    }
}

// ---------------------------------------------------------------------------
// RMSNorm: one block per row of (512, DMODEL)
// ---------------------------------------------------------------------------
__global__ __launch_bounds__(256)
void rmsnorm_kernel(const float* __restrict__ x, const float* __restrict__ w,
                    float* __restrict__ out, int D)
{
    __shared__ float red[8];
    const int row = blockIdx.x;
    const float* xr = x + (size_t)row * D;
    float s = 0.0f;
    for (int i = threadIdx.x; i < D; i += 256) { float v = xr[i]; s += v * v; }
    for (int off = 16; off > 0; off >>= 1) s += __shfl_down(s, off, 32);
    const int lane = threadIdx.x & 31, wv = threadIdx.x >> 5;
    if (lane == 0) red[wv] = s;
    __syncthreads();
    if (wv == 0) {
        float t = (lane < 8) ? red[lane] : 0.0f;
        for (int off = 4; off > 0; off >>= 1) t += __shfl_down(t, off, 32);
        if (lane == 0) red[0] = t;
    }
    __syncthreads();
    const float inv = rsqrtf(red[0] / (float)D + 1e-5f);
    for (int i = threadIdx.x; i < D; i += 256)
        out[(size_t)row * D + i] = xr[i] * inv * w[i];
}

// ---------------------------------------------------------------------------
// Causal depthwise conv (window 4) + SiLU over u = xz[:, :DINNER]
// ---------------------------------------------------------------------------
__global__ __launch_bounds__(256)
void conv_silu_kernel(const float* __restrict__ xz, const float* __restrict__ cw,
                      const float* __restrict__ cb, float* __restrict__ uc)
{
    int idx = blockIdx.x * 256 + threadIdx.x;      // (b*L + l)*DINNER + d
    if (idx >= BATCH * SEQL * DINNER) return;
    const int d  = idx & (DINNER - 1);
    const int bl = idx >> 11;                      // DINNER == 2048
    const int l  = bl & (SEQL - 1);
    float s = cb[d];
    #pragma unroll
    for (int j = 0; j < 4; ++j) {
        int ls = l - 3 + j;
        if (ls >= 0)
            s += cw[j * DINNER + d] * xz[(size_t)(bl - 3 + j) * (2 * DINNER) + d];
    }
    float sig = 1.0f / (1.0f + __expf(-s));
    uc[(size_t)bl * DINNER + d] = s * sig;         // silu
}

// ---------------------------------------------------------------------------
// Selective scan: one thread per (batch, channel); state in registers.
// Fuses dA/dBu formation, the recurrence, D-skip and the SiLU(z) gate.
// ---------------------------------------------------------------------------
__global__ __launch_bounds__(256)
void scan_kernel(const float* __restrict__ dt, const float* __restrict__ uc,
                 const float* __restrict__ xz, const float* __restrict__ dbc,
                 const float* __restrict__ A_log, const float* __restrict__ Dskip,
                 float* __restrict__ ys)
{
    int idx = blockIdx.x * 256 + threadIdx.x;      // b*DINNER + d
    if (idx >= BATCH * DINNER) return;
    const int d = idx & (DINNER - 1);
    const int b = idx >> 11;

    float a_c[DSTATE], h[DSTATE];
    #pragma unroll
    for (int n = 0; n < DSTATE; ++n) {
        a_c[n] = -__expf(A_log[d * DSTATE + n]);
        h[n]   = 0.0f;
    }
    const float dsk = Dskip[d];

    for (int l = 0; l < SEQL; ++l) {
        const int r = b * SEQL + l;
        const float dtv = dt[(size_t)r * DINNER + d];
        const float uv  = uc[(size_t)r * DINNER + d];
        const float zv  = xz[(size_t)r * (2 * DINNER) + DINNER + d];
        const float du  = dtv * uv;
        const float* bm = dbc + (size_t)r * (DTRANK + 2 * DSTATE) + DTRANK;
        const float* cm = bm + DSTATE;
        float y = 0.0f;
        #pragma unroll
        for (int n = 0; n < DSTATE; ++n) {
            h[n] = __expf(dtv * a_c[n]) * h[n] + du * bm[n];
            y   += h[n] * cm[n];
        }
        y += dsk * uv;
        const float sig = 1.0f / (1.0f + __expf(-zv));
        ys[(size_t)r * DINNER + d] = y * (zv * sig);
    }
}

// ---------------------------------------------------------------------------
// Launcher
// ---------------------------------------------------------------------------
extern "C" void kernel_launch(void* const* d_in, const int* in_sizes, int n_in,
                              void* d_out, int out_size, void* d_ws, size_t ws_size,
                              hipStream_t stream)
{
    const float* x          = (const float*)d_in[0];
    const float* enc_w      = (const float*)d_in[1];
    const float* enc_b      = (const float*)d_in[2];
    const float* dec_w      = (const float*)d_in[3];
    const float* dec_b      = (const float*)d_in[4];
    const float* norm_w     = (const float*)d_in[5];
    const float* in_proj_w  = (const float*)d_in[6];
    const float* conv_w     = (const float*)d_in[7];
    const float* conv_b     = (const float*)d_in[8];
    const float* x_proj_w   = (const float*)d_in[9];
    const float* dt_proj_w  = (const float*)d_in[10];
    const float* dt_proj_b  = (const float*)d_in[11];
    const float* A_log      = (const float*)d_in[12];
    const float* Dskip      = (const float*)d_in[13];
    const float* out_proj_w = (const float*)d_in[14];
    float* out = (float*)d_out;

    const int ROWS = BATCH * SEQL;   // 512
    float* ws   = (float*)d_ws;
    float* lat  = ws; ws += (size_t)ROWS * DMODEL;
    float* xn   = ws; ws += (size_t)ROWS * DMODEL;
    float* xz   = ws; ws += (size_t)ROWS * 2 * DINNER;
    float* uc   = ws; ws += (size_t)ROWS * DINNER;
    float* dbc  = ws; ws += (size_t)ROWS * (DTRANK + 2 * DSTATE);
    float* dtb  = ws; ws += (size_t)ROWS * DINNER;
    float* ysb  = ws; ws += (size_t)ROWS * DINNER;
    float* hout = ws; ws += (size_t)ROWS * DMODEL;

    dim3 blk(256);
    const int MT = ROWS / TM;        // 4 M tiles

    // 1) encoder: lat = x @ enc_w^T + enc_b          (512x32768)(32768x1024)
    gemm_bf16_wmma<false><<<dim3(MT, DMODEL / TN), blk, 0, stream>>>(
        x, FLATHW, enc_w, enc_b, nullptr, lat, ROWS, DMODEL, FLATHW, 0, 0);

    // 2) rmsnorm
    rmsnorm_kernel<<<ROWS, blk, 0, stream>>>(lat, norm_w, xn, DMODEL);

    // 3) in_proj: xz = xn @ in_proj_w^T              (512x1024)(1024x4096)
    gemm_bf16_wmma<false><<<dim3(MT, (2 * DINNER) / TN), blk, 0, stream>>>(
        xn, DMODEL, in_proj_w, nullptr, nullptr, xz, ROWS, 2 * DINNER, DMODEL, 0, 0);

    // 4) causal depthwise conv + silu
    conv_silu_kernel<<<(ROWS * DINNER) / 256, blk, 0, stream>>>(xz, conv_w, conv_b, uc);

    // 5) x_proj: dbc = u @ x_proj_w^T                (512x2048)(2048x96)
    gemm_bf16_wmma<true><<<dim3(MT, 1), blk, 0, stream>>>(
        uc, DINNER, x_proj_w, nullptr, nullptr, dbc,
        ROWS, DTRANK + 2 * DSTATE, DINNER, 0, 0);

    // 6) dt = softplus(dt_r @ dt_proj_w^T + b)       (512x64)(64x2048)
    gemm_bf16_wmma<false><<<dim3(MT, DINNER / TN), blk, 0, stream>>>(
        dbc, DTRANK + 2 * DSTATE, dt_proj_w, dt_proj_b, nullptr, dtb,
        ROWS, DINNER, DTRANK, 2, 0);

    // 7) selective scan (+ D-skip + silu(z) gate)
    scan_kernel<<<(BATCH * DINNER) / 256, blk, 0, stream>>>(
        dtb, uc, xz, dbc, A_log, Dskip, ysb);

    // 8) out_proj + residual: hout = lat + ys @ out_proj_w^T
    gemm_bf16_wmma<false><<<dim3(MT, DMODEL / TN), blk, 0, stream>>>(
        ysb, DINNER, out_proj_w, nullptr, lat, hout, ROWS, DMODEL, DINNER, 0, 0);

    // 9) decoder + sigmoid + [:, :-1] slice straight into d_out
    gemm_bf16_wmma<false><<<dim3(MT, FLATHW / TN), blk, 0, stream>>>(
        hout, DMODEL, dec_w, dec_b, nullptr, out, ROWS, FLATHW, DMODEL, 1, 1);
}
